// GraphConvEmbedding_84679575208612
// MI455X (gfx1250) — matrix operand order repaired and checked by reading, not compiled, not run on hardware
//
#include <hip/hip_runtime.h>
#include <hip/hip_bf16.h>
#include <math.h>

#define B_ 256
#define N_ 256
#define D_ 128
#define K_ 10
#define NUM_NODES_ (B_ * N_)          // 65536
#define E_ (B_ * 8192)                // 2097152
#define HALF_CH_ 64
#define OUT_CH_ 128
#define KSIZE_ 5

typedef __attribute__((ext_vector_type(16))) __bf16 v16bf;
typedef __attribute__((ext_vector_type(8)))  __bf16 v8bf;
typedef __attribute__((ext_vector_type(8)))  float  v8f;

// ---------------- utility ----------------
__global__ void zero_kernel(float* __restrict__ p, int n) {
    int i = blockIdx.x * blockDim.x + threadIdx.x;
    if (i < n) p[i] = 0.0f;
}

// ---------------- degrees ----------------
__global__ void degree_kernel(const int* __restrict__ src, const int* __restrict__ dst,
                              float* __restrict__ degO, float* __restrict__ degI) {
    int e = blockIdx.x * blockDim.x + threadIdx.x;
    if (e >= E_) return;
    atomicAdd(&degO[src[e]], 1.0f);
    atomicAdd(&degI[dst[e]], 1.0f);
}

__global__ void isqrt_kernel(const float* __restrict__ degO, const float* __restrict__ degI,
                             float* __restrict__ odis, float* __restrict__ idis) {
    int n = blockIdx.x * blockDim.x + threadIdx.x;
    if (n >= NUM_NODES_) return;
    odis[n] = rsqrtf(fmaxf(degO[n], 1.0f));
    idis[n] = rsqrtf(fmaxf(degI[n], 1.0f));
}

// ---------------- edge gather + scatter-add (L2-resident atomics) ----------------
__global__ void scatter_kernel(const float* __restrict__ H, const int* __restrict__ src,
                               const int* __restrict__ dst, const float* __restrict__ odis,
                               float* __restrict__ agg) {
    int tid  = blockIdx.x * blockDim.x + threadIdx.x;   // E_*32 = 67,108,864 < 2^31
    int e    = tid >> 5;
    int lane = tid & 31;
    if (e >= E_) return;
    int s = src[e], d = dst[e];
    float sc = odis[s];
    const float4 v = *(const float4*)(H + (size_t)s * D_ + lane * 4);
    float* ar = agg + (size_t)d * D_ + lane * 4;
    atomicAdd(ar + 0, v.x * sc);
    atomicAdd(ar + 1, v.y * sc);
    atomicAdd(ar + 2, v.z * sc);
    atomicAdd(ar + 3, v.w * sc);
}

// ---------------- W prep: WT_bf16[n][k] = (bf16) W[k][n] ----------------
__global__ void wprep_kernel(const float* __restrict__ W, __bf16* __restrict__ WT) {
    int t = blockIdx.x * blockDim.x + threadIdx.x;        // 16384
    int n = t >> 7, k = t & 127;
    WT[t] = (__bf16)W[k * D_ + n];
}

// ---------------- WMMA GEMM: H = relu( (rowscale * A) @ W + bias ) ----------------
// Block: 256 threads = 8 waves, computes a 64x128 output tile.
// LDS: A tile 64x128 bf16 (row-scaled), WT 128x128 bf16 (column-major of W).
// Each wave: 1 N-tile (16 cols) x 4 M-tiles, B fragments VGPR-resident -> 16 WMMAs.
__global__ __launch_bounds__(256) void gemm_wmma_kernel(
        const float* __restrict__ A, const __bf16* __restrict__ WT,
        const float* __restrict__ bias, const float* __restrict__ rowscale,
        float* __restrict__ Hout) {
    __shared__ __attribute__((aligned(32))) __bf16 Alds[64 * D_];    // 16 KB
    __shared__ __attribute__((aligned(32))) __bf16 Wlds[D_ * D_];    // 32 KB

    const int t = threadIdx.x;
    const int mbase = blockIdx.x * 64;

    // ---- stage A tile: f32 global -> rowscaled bf16 LDS (pairs, packed b32 stores)
    const float2* A2 = (const float2*)(A + (size_t)mbase * D_);
#pragma unroll
    for (int i = 0; i < 16; ++i) {
        int idx = t + i * 256;                 // 0..4095 float2 pairs
        int row = idx >> 6;                    // 64 pairs per row
        float sc = rowscale[mbase + row];
        float2 v = A2[idx];
        union { __bf16 h[2]; unsigned u; } pk;
        pk.h[0] = (__bf16)(v.x * sc);
        pk.h[1] = (__bf16)(v.y * sc);
        ((unsigned*)Alds)[idx] = pk.u;
    }
    // ---- stage WT: bf16 global -> LDS via 16B chunks
    const uint4* Wg = (const uint4*)WT;        // 2048 x 16B
    uint4* Wl = (uint4*)Wlds;
#pragma unroll
    for (int i = 0; i < 8; ++i)
        Wl[t + i * 256] = Wg[t + i * 256];
    __syncthreads();

    const int lane = t & 31;
    const int hi = lane >> 4;
    const int l  = lane & 15;
    const int wv = t >> 5;                     // N tile id (0..7)
    const int ncol = wv * 16 + l;

    // ---- B fragments (VGPR resident): b[i] = WT[ncol][kc*32 + 16*hi + i], contiguous
    v16bf bfrag[4];
#pragma unroll
    for (int kc = 0; kc < 4; ++kc)
        bfrag[kc] = *(const v16bf*)&Wlds[ncol * D_ + kc * 32 + hi * 16];

    v8f acc[4] = {v8f{}, v8f{}, v8f{}, v8f{}};
#pragma unroll
    for (int m = 0; m < 4; ++m) {
        const __bf16* arow = &Alds[(m * 16 + l) * D_ + hi * 16];
#pragma unroll
        for (int kc = 0; kc < 4; ++kc) {
            // a[0..15] = K kb+16hi..+15 (contiguous)
            v8bf alo = *(const v8bf*)(arow + kc * 32);
            v8bf ahi = *(const v8bf*)(arow + kc * 32 + 8);
            v16bf a = __builtin_shufflevector(alo, ahi,
                        0, 1, 2, 3, 4, 5, 6, 7, 8, 9, 10, 11, 12, 13, 14, 15);
            acc[m] = __builtin_amdgcn_wmma_f32_16x16x32_bf16(
                         false, a, false, bfrag[kc], (short)0, acc[m], false, false);
        }
    }

    float bn = bias[ncol];
#pragma unroll
    for (int m = 0; m < 4; ++m)
#pragma unroll
        for (int i = 0; i < 8; ++i) {
            int grow = mbase + m * 16 + hi * 8 + i;     // C/D: M = i + 8*hi
            float v = acc[m][i] + bn;
            Hout[(size_t)grow * D_ + ncol] = v > 0.0f ? v : 0.0f;
        }
}

// ---------------- keys = row max ----------------
__global__ void rowmax_kernel(const float* __restrict__ H, float* __restrict__ keys) {
    int n = blockIdx.x * blockDim.x + threadIdx.x;
    if (n >= NUM_NODES_) return;
    const float* row = H + (size_t)n * D_;
    float m = row[0];
#pragma unroll 4
    for (int i = 1; i < D_; ++i) m = fmaxf(m, row[i]);
    keys[n] = m;
}

// ---------------- top-K per graph (destroys keys; keys recomputed each launch) ----------------
__global__ void topk_kernel(float* __restrict__ keys, int* __restrict__ idx) {
    int b = blockIdx.x * blockDim.x + threadIdx.x;
    if (b >= B_) return;
    float* kb = keys + b * N_;
    for (int k = 0; k < K_; ++k) {
        float best = kb[0];
        int bi = 0;
        for (int i = 1; i < N_; ++i) {
            float v = kb[i];
            if (v > best) { best = v; bi = i; }           // strict > : first index on ties
        }
        idx[b * K_ + k] = bi;
        kb[bi] = -__builtin_inff();
    }
}

// ---------------- bitonic sort of the 2560 selected rows (ascending) ----------------
__global__ void sortrows_kernel(const float* __restrict__ H, const int* __restrict__ idx,
                                float* __restrict__ pooled) {
    __shared__ float s[D_];
    int row = blockIdx.x;                 // 0 .. B_*K_-1
    int b = row / K_;
    int node = b * N_ + idx[row];
    int t = threadIdx.x;                  // 128 threads
    s[t] = H[(size_t)node * D_ + t];
    __syncthreads();
    for (int k = 2; k <= D_; k <<= 1) {
        for (int j = k >> 1; j > 0; j >>= 1) {
            int ixj = t ^ j;
            if (ixj > t) {
                bool up = ((t & k) == 0);
                float x = s[t], y = s[ixj];
                if ((x > y) == up) { s[t] = y; s[ixj] = x; }
            }
            __syncthreads();
        }
    }
    pooled[(size_t)row * D_ + t] = s[t];
}

// ---------------- conv1 (stride-D pointwise): out1[b][c][k] ----------------
__global__ void conv1_kernel(const float* __restrict__ pooled, const float* __restrict__ w,
                             const float* __restrict__ bias, float* __restrict__ out1) {
    int t = blockIdx.x * blockDim.x + threadIdx.x;        // B_*64*10
    if (t >= B_ * HALF_CH_ * K_) return;
    int k = t % K_;
    int c = (t / K_) % HALF_CH_;
    int b = t / (K_ * HALF_CH_);
    const float* p  = pooled + (size_t)(b * K_ + k) * D_;
    const float* wc = w + (size_t)c * D_;
    float acc = bias[c];
#pragma unroll 4
    for (int d = 0; d < D_; ++d) acc += p[d] * wc[d];
    out1[t] = acc > 0.0f ? acc : 0.0f;
}

// ---------------- maxpool (window 2, stride 2) ----------------
__global__ void maxpool_kernel(const float* __restrict__ out1, float* __restrict__ pool2) {
    int t = blockIdx.x * blockDim.x + threadIdx.x;        // B_*64*5
    if (t >= B_ * HALF_CH_ * 5) return;
    int u = t % 5;
    int bc = t / 5;
    float a = out1[bc * K_ + 2 * u];
    float b = out1[bc * K_ + 2 * u + 1];
    pool2[t] = fmaxf(a, b);
}

// ---------------- conv2 (64ch x 5 taps -> 128ch, length 1) ----------------
__global__ void conv2_kernel(const float* __restrict__ pool2, const float* __restrict__ w,
                             const float* __restrict__ bias, float* __restrict__ out) {
    int t = blockIdx.x * blockDim.x + threadIdx.x;        // B_*128
    if (t >= B_ * OUT_CH_) return;
    int o = t % OUT_CH_;
    int b = t / OUT_CH_;
    float acc = bias[o];
    const float* pb = pool2 + (size_t)b * HALF_CH_ * 5;
    const float* wo = w + (size_t)o * HALF_CH_ * KSIZE_;
#pragma unroll 4
    for (int c = 0; c < HALF_CH_; ++c)
#pragma unroll
        for (int s = 0; s < KSIZE_; ++s)
            acc += pb[c * 5 + s] * wo[c * KSIZE_ + s];
    out[t] = acc > 0.0f ? acc : 0.0f;
}

extern "C" void kernel_launch(void* const* d_in, const int* in_sizes, int n_in,
                              void* d_out, int out_size, void* d_ws, size_t ws_size,
                              hipStream_t stream) {
    const float* feats = (const float*)d_in[0];
    const int*   src   = (const int*)d_in[1];
    const int*   dst   = (const int*)d_in[2];
    const float* W0    = (const float*)d_in[3];
    const float* b0    = (const float*)d_in[4];
    const float* W1    = (const float*)d_in[5];
    const float* b1    = (const float*)d_in[6];
    const float* W2    = (const float*)d_in[7];
    const float* b2    = (const float*)d_in[8];
    const float* c1w   = (const float*)d_in[9];
    const float* c1b   = (const float*)d_in[10];
    const float* c2w   = (const float*)d_in[11];
    const float* c2b   = (const float*)d_in[12];
    float* out = (float*)d_out;

    char* wp = (char*)d_ws;
    auto carve = [&](size_t bytes) -> void* {
        void* p = (void*)wp;
        wp += (bytes + 255) & ~(size_t)255;
        return p;
    };
    float*  degO   = (float*)carve((size_t)NUM_NODES_ * 4);
    float*  degI   = (float*)carve((size_t)NUM_NODES_ * 4);
    float*  odis   = (float*)carve((size_t)NUM_NODES_ * 4);
    float*  idis   = (float*)carve((size_t)NUM_NODES_ * 4);
    float*  keys   = (float*)carve((size_t)NUM_NODES_ * 4);
    int*    idx    = (int*)  carve((size_t)B_ * K_ * 4);
    float*  pooled = (float*)carve((size_t)B_ * K_ * D_ * 4);
    float*  out1   = (float*)carve((size_t)B_ * HALF_CH_ * K_ * 4);
    float*  pool2  = (float*)carve((size_t)B_ * HALF_CH_ * 5 * 4);
    __bf16* WTbf   = (__bf16*)carve((size_t)D_ * D_ * 2);
    float*  h      = (float*)carve((size_t)NUM_NODES_ * D_ * 4);
    float*  agg    = (float*)carve((size_t)NUM_NODES_ * D_ * 4);

    const int ND = NUM_NODES_ * D_;

    // degrees + normalization
    zero_kernel<<<(NUM_NODES_ + 255) / 256, 256, 0, stream>>>(degO, NUM_NODES_);
    zero_kernel<<<(NUM_NODES_ + 255) / 256, 256, 0, stream>>>(degI, NUM_NODES_);
    degree_kernel<<<E_ / 256, 256, 0, stream>>>(src, dst, degO, degI);
    isqrt_kernel<<<NUM_NODES_ / 256, 256, 0, stream>>>(degO, degI, odis, idis);

    // 3 graph-conv layers: W prep -> scatter into agg (L2-resident) -> WMMA GEMM into h
    auto layer = [&](const float* hin, const float* W, const float* bias) {
        wprep_kernel<<<(D_ * D_) / 256, 256, 0, stream>>>(W, WTbf);
        zero_kernel<<<ND / 256, 256, 0, stream>>>(agg, ND);
        scatter_kernel<<<(E_ * 32) / 256, 256, 0, stream>>>(hin, src, dst, odis, agg);
        gemm_wmma_kernel<<<NUM_NODES_ / 64, 256, 0, stream>>>(agg, WTbf, bias, idis, h);
    };
    layer(feats, W0, b0);
    layer(h,     W1, b1);
    layer(h,     W2, b2);

    // keys = row max, per-graph top-K, sort only the selected 2560 rows
    rowmax_kernel<<<NUM_NODES_ / 256, 256, 0, stream>>>(h, keys);
    topk_kernel<<<1, 256, 0, stream>>>(keys, idx);
    sortrows_kernel<<<B_ * K_, D_, 0, stream>>>(h, idx, pooled);

    // head: conv1 -> maxpool -> conv2
    conv1_kernel<<<(B_ * HALF_CH_ * K_ + 255) / 256, 256, 0, stream>>>(pooled, c1w, c1b, out1);
    maxpool_kernel<<<(B_ * HALF_CH_ * 5 + 255) / 256, 256, 0, stream>>>(out1, pool2);
    conv2_kernel<<<(B_ * OUT_CH_ + 255) / 256, 256, 0, stream>>>(pool2, c2w, c2b, out);
}
